// FOGCNConv_45518063403582
// MI455X (gfx1250) — matrix operand order repaired and checked by reading, not compiled, not run on hardware
//
#include <hip/hip_runtime.h>
#include <hip/hip_bf16.h>

typedef __attribute__((ext_vector_type(16))) _Float16 v16h;
typedef __attribute__((ext_vector_type(8)))  float    v8f;
typedef __attribute__((ext_vector_type(4)))  float    v4f;

#define N_NODES   20000
#define N_EDGES   640000
#define NUM_C     64
#define NUM_F     128
#define NF        (N_NODES * NUM_F)          // 2,560,000 output elems
#define N_TILES   (N_EDGES / 16)             // 40000 16-edge wave tiles
#define MAIN_BLOCKS 1250                     // 1250*8 waves * 4 iters = 40000
#define BFRAG_HALVES (8 * 2 * 32 * 16)       // 8192 halves = 16 KB

// ---------------------------------------------------------------------------
// K1: softmax over counts axis (per feature column), then pack weight into
// the per-lane WMMA B-fragment layout for V_WMMA_F32_16X16X32_F16.
// B (32x16, 16-bit): lanes 0-15 hold K=0..7,16..23, lanes 16-31
// hold K=8..15,24..31; column N = lane & 15. Two K-fragments (q) cover K=0..63.
// bfrag element index: (((t*2 + q)*32 + lane)*16 + j), t = N-tile 0..7.
// ---------------------------------------------------------------------------
__global__ void fogcn_softmax_pack(const float* __restrict__ importance,
                                   _Float16* __restrict__ bfrag) {
    __shared__ float w[NUM_C * NUM_F];       // 32 KB of the 320 KB LDS
    int f = threadIdx.x;                     // 128 threads, one per feature
    float m = -1e30f;
    for (int c = 0; c < NUM_C; ++c) m = fmaxf(m, importance[c * NUM_F + f]);
    float s = 0.f;
    for (int c = 0; c < NUM_C; ++c) s += __expf(importance[c * NUM_F + f] - m);
    float inv = 1.f / s;
    for (int c = 0; c < NUM_C; ++c)
        w[c * NUM_F + f] = __expf(importance[c * NUM_F + f] - m) * inv;
    __syncthreads();
    for (int e = threadIdx.x; e < BFRAG_HALVES; e += blockDim.x) {
        int t    = e >> 10;
        int q    = (e >> 9) & 1;
        int lane = (e >> 4) & 31;
        int j    = e & 15;
        int K    = ((lane & 16) ? 8 : 0) + 32 * q + (j < 8 ? j : j + 8);
        int N    = 16 * t + (lane & 15);
        bfrag[e] = (_Float16)w[K * NUM_F + N];
    }
}

// ---------------------------------------------------------------------------
// K2: zero the two f32 accumulators (msgs -> d_out, node_score -> ws)
// ---------------------------------------------------------------------------
__global__ void fogcn_zero(float* __restrict__ out, float* __restrict__ score) {
    int i = blockIdx.x * blockDim.x + threadIdx.x;
    if (i < NF) out[i] = 0.f;
    else if (i < 2 * NF) score[i - NF] = 0.f;
}

// ---------------------------------------------------------------------------
// K3: fused  edge_score = cnt @ weight  (WMMA f16->f32, K=64)
//            msgs = embedding[src] * edge_score  (L2-resident gather)
//            segment_sum via global_atomic_add_f32 into out / score.
// One wave per 16-edge tile. B fragments staged in LDS per block.
// cnt is streamed once -> non-temporal loads keep L2 for reused data.
// All gather/scatter offsets kept in 32 bits -> saddr+voffset addressing.
// Epilogue split: issue all 8 gathers (covers WMMA->VALU hazard + latency),
// then consume with mul + dual atomics.
// ---------------------------------------------------------------------------
__global__ void fogcn_main(const float*   __restrict__ cnt,
                           const float*   __restrict__ embedding,
                           const int*     __restrict__ src,
                           const int*     __restrict__ dst,
                           const _Float16* __restrict__ bfrag,
                           float* __restrict__ out,
                           float* __restrict__ score) {
    __shared__ __align__(32) _Float16 bl[BFRAG_HALVES];   // 16 KB
    {   // stage packed B fragments into LDS (uint4 = 8 halves per copy)
        const uint4* g = (const uint4*)bfrag;
        uint4* l = (uint4*)bl;
        for (int i = threadIdx.x; i < BFRAG_HALVES / 8; i += blockDim.x)
            l[i] = g[i];
    }
    __syncthreads();

    const int lane  = threadIdx.x & 31;
    const int wave  = threadIdx.x >> 5;
    const int stride = gridDim.x * 8;        // wave-tiles per sweep
    const int mrow  = lane & 15;
    const int kb    = (lane & 16) ? 8 : 0;   // A-frag K base (ISA 7.12.2)
    const int nloc  = lane & 15;
    const int mbase = (lane & 16) ? 8 : 0;   // C-frag M base

    for (int tile = blockIdx.x * 8 + wave; tile < N_TILES; tile += stride) {
        int nt = tile + stride;              // prefetch next tile's cnt rows
        if (nt < N_TILES)
            __builtin_prefetch(cnt + (size_t)nt * 16 * NUM_C, 0, 1);

        const int tb = tile * 16;
        const float* arow = cnt + (size_t)(tb + mrow) * NUM_C;

        // ---- build A fragments (f32 -> f16), non-temporal streaming ----
        v4f q0 = __builtin_nontemporal_load((const v4f*)(arow + kb));
        v4f q1 = __builtin_nontemporal_load((const v4f*)(arow + kb + 4));
        v4f q2 = __builtin_nontemporal_load((const v4f*)(arow + kb + 16));
        v4f q3 = __builtin_nontemporal_load((const v4f*)(arow + kb + 20));
        v4f r0 = __builtin_nontemporal_load((const v4f*)(arow + kb + 32));
        v4f r1 = __builtin_nontemporal_load((const v4f*)(arow + kb + 36));
        v4f r2 = __builtin_nontemporal_load((const v4f*)(arow + kb + 48));
        v4f r3 = __builtin_nontemporal_load((const v4f*)(arow + kb + 52));
        v16h a0, a1;
        a0[0]=(_Float16)q0.x; a0[1]=(_Float16)q0.y; a0[2]=(_Float16)q0.z; a0[3]=(_Float16)q0.w;
        a0[4]=(_Float16)q1.x; a0[5]=(_Float16)q1.y; a0[6]=(_Float16)q1.z; a0[7]=(_Float16)q1.w;
        a0[8]=(_Float16)q2.x; a0[9]=(_Float16)q2.y; a0[10]=(_Float16)q2.z; a0[11]=(_Float16)q2.w;
        a0[12]=(_Float16)q3.x; a0[13]=(_Float16)q3.y; a0[14]=(_Float16)q3.z; a0[15]=(_Float16)q3.w;
        a1[0]=(_Float16)r0.x; a1[1]=(_Float16)r0.y; a1[2]=(_Float16)r0.z; a1[3]=(_Float16)r0.w;
        a1[4]=(_Float16)r1.x; a1[5]=(_Float16)r1.y; a1[6]=(_Float16)r1.z; a1[7]=(_Float16)r1.w;
        a1[8]=(_Float16)r2.x; a1[9]=(_Float16)r2.y; a1[10]=(_Float16)r2.z; a1[11]=(_Float16)r2.w;
        a1[12]=(_Float16)r3.x; a1[13]=(_Float16)r3.y; a1[14]=(_Float16)r3.z; a1[15]=(_Float16)r3.w;

        // ---- per-lane edge indices for the 8 rows this lane's C covers ----
        int svr[8], dvr[8];
#pragma unroll
        for (int r = 0; r < 8; ++r) {
            svr[r] = src[tb + mbase + r];
            dvr[r] = dst[tb + mbase + r];
        }

        // ---- 8 N-tiles of 16 features; unroll 2 to bound VGPR pressure ----
#pragma unroll 2
        for (int t = 0; t < 8; ++t) {
            v16h b0 = *(const v16h*)(bl + ((t * 2 + 0) * 32 + lane) * 16);
            v16h b1 = *(const v16h*)(bl + ((t * 2 + 1) * 32 + lane) * 16);
            v8f c = {};
            c = __builtin_amdgcn_wmma_f32_16x16x32_f16(false, a0, false, b0,
                                                       (short)0, c, false, false);
            c = __builtin_amdgcn_wmma_f32_16x16x32_f16(false, a1, false, b1,
                                                       (short)0, c, false, false);
            const unsigned nglob = (unsigned)(t * 16 + nloc);

            // (a) issue all 8 gathers: independent of c, overlaps WMMA latency
            float ev[8];
#pragma unroll
            for (int r = 0; r < 8; ++r)
                ev[r] = embedding[(((unsigned)svr[r]) << 7) | nglob];

            // (b) consume: mul + paired atomics (L2-resident accumulators)
#pragma unroll
            for (int r = 0; r < 8; ++r) {
                float s = c[r];
                unsigned oo = ((unsigned)dvr[r] << 7) | nglob;
                atomicAdd(out + oo, ev[r] * s);
                atomicAdd(score + oo, s);
            }
        }
    }
}

// ---------------------------------------------------------------------------
// K4: normalize
// ---------------------------------------------------------------------------
__global__ void fogcn_div(float* __restrict__ out, const float* __restrict__ score) {
    int i = blockIdx.x * blockDim.x + threadIdx.x;
    if (i < NF) out[i] = out[i] / score[i];
}

extern "C" void kernel_launch(void* const* d_in, const int* in_sizes, int n_in,
                              void* d_out, int out_size, void* d_ws, size_t ws_size,
                              hipStream_t stream) {
    const float* cnt        = (const float*)d_in[0];
    const float* embedding  = (const float*)d_in[1];
    const float* importance = (const float*)d_in[2];
    const int*   src        = (const int*)d_in[3];
    const int*   dst        = (const int*)d_in[4];
    float* out = (float*)d_out;

    char* ws = (char*)d_ws;
    _Float16* bfrag = (_Float16*)ws;                 // 16 KB packed B fragments
    float*    score = (float*)(ws + 16384);          // N*F f32 node_score accum

    fogcn_softmax_pack<<<1, 128, 0, stream>>>(importance, bfrag);
    fogcn_zero<<<(2 * NF + 255) / 256, 256, 0, stream>>>(out, score);
    fogcn_main<<<MAIN_BLOCKS, 256, 0, stream>>>(cnt, embedding, src, dst,
                                                bfrag, out, score);
    fogcn_div<<<(NF + 255) / 256, 256, 0, stream>>>(out, score);
}
